// CausalSelfAttention_17008070492992
// MI455X (gfx1250) — compile-verified, matrix operand order checked by reading, not run on hardware
//
#include <hip/hip_runtime.h>

// ---------------------------------------------------------------------------
// Causal self-attention for MI455X (gfx1250, wave32, WMMA).
// B=2 S=2048 D=768 H=12 HD=64.  Compute-bound (~32 GFLOP vs ~35MB traffic),
// so all GEMMs run on v_wmma_f32_16x16x32_bf16 (f32 accumulate).
// ---------------------------------------------------------------------------

#define BB   2
#define SSQ  2048
#define DDM  768
#define HHN  12
#define HDIM 64
#define MMR  (BB * SSQ)   // 4096 flat rows

typedef __attribute__((ext_vector_type(16))) __bf16        v16bf;
typedef __attribute__((ext_vector_type(8)))  float         v8f;
typedef __attribute__((ext_vector_type(4)))  unsigned int  v4u;

union FragBf {
    v16bf v;
    v4u   u[2];
};

__device__ __forceinline__ unsigned short f2bf_bits(float f) {
    unsigned int u = __float_as_uint(f);
    u += 0x7fffu + ((u >> 16) & 1u);          // round-to-nearest-even
    return (unsigned short)(u >> 16);
}

// XOR-butterfly across the 32 lanes (masks 1,2,4,8 stay inside each 16-lane
// half, matching the WMMA C-matrix row split).
#define SWZF(val, m) \
    __int_as_float(__builtin_amdgcn_ds_swizzle(__float_as_int(val), (((m) << 10) | 0x1f)))

__device__ __forceinline__ v8f wmma_bf16(const v16bf& a, const v16bf& b, const v8f& c) {
    return __builtin_amdgcn_wmma_f32_16x16x32_bf16(false, a, false, b, (short)0, c, false, false);
}

// ---------------------------------------------------------------------------
// fp32 -> bf16 bulk convert
// ---------------------------------------------------------------------------
__global__ void k_f32_to_bf16(const float* __restrict__ in,
                              unsigned short* __restrict__ out, int n) {
    int i = blockIdx.x * blockDim.x + threadIdx.x;
    if (i < n) out[i] = f2bf_bits(in[i]);
}

// fp32 [k][n] -> bf16 transposed [n][k]   (weights, 768x768)
__global__ void k_transpose_bf16(const float* __restrict__ in,
                                 unsigned short* __restrict__ out) {
    int idx = blockIdx.x * blockDim.x + threadIdx.x;     // over DDM*DDM
    int k = idx % DDM;
    int n = idx / DDM;
    out[idx] = f2bf_bits(in[(size_t)k * DDM + n]);
}

// ---------------------------------------------------------------------------
// One wave computes a 64(M) x 64(N) f32 tile: A [64][DDM] row-major bf16,
// Bt [64][DDM] row-major bf16 (= columns of original W).  Per k-step of 32:
// 4 A frags + 4 B frags (all contiguous b128 loads) -> 16 WMMAs.
// ---------------------------------------------------------------------------
__device__ __forceinline__ void wave_gemm_64x64(const unsigned short* __restrict__ A,
                                                const unsigned short* __restrict__ Bt,
                                                v8f (&acc)[4][4], int l16, int half) {
#pragma unroll 1
    for (int k0 = 0; k0 < DDM; k0 += 32) {
        const int c0 = k0 + half * 8;     // lane<16: K {0..7,16..23}; lane>=16: {8..15,24..31}
        FragBf a[4];
#pragma unroll
        for (int i = 0; i < 4; i++) {
            const unsigned short* p = A + (size_t)(i * 16 + l16) * DDM + c0;
            a[i].u[0] = *(const v4u*)(p);
            a[i].u[1] = *(const v4u*)(p + 16);
        }
#pragma unroll
        for (int j = 0; j < 4; j++) {
            FragBf b;
            const unsigned short* p = Bt + (size_t)(j * 16 + l16) * DDM + c0;
            b.u[0] = *(const v4u*)(p);
            b.u[1] = *(const v4u*)(p + 16);
#pragma unroll
            for (int i = 0; i < 4; i++)
                acc[i][j] = wmma_bf16(a[i].v, b.v, acc[i][j]);
        }
    }
}

// ---------------------------------------------------------------------------
// Fused QKV projection.  grid = (MMR/256, H, 3), block = 128 (4 waves).
// z selects {Q,K,V}.  Q,K stored [B,H,S,HD] bf16; V stored transposed
// [B,H,HD,S] bf16 (so the P*V B-operand is contiguous, and the transposed
// store packs 8 consecutive s per lane into one b128 store).
// ---------------------------------------------------------------------------
__global__ __launch_bounds__(128) void k_qkv(
    const unsigned short* __restrict__ xb,
    const unsigned short* __restrict__ wqt, const unsigned short* __restrict__ wkt,
    const unsigned short* __restrict__ wvt,
    const float* __restrict__ bq, const float* __restrict__ bk, const float* __restrict__ bv,
    unsigned short* __restrict__ qb, unsigned short* __restrict__ kb,
    unsigned short* __restrict__ vtb) {
    const int wave = threadIdx.x >> 5;
    const int lane = threadIdx.x & 31;
    const int half = lane >> 4;
    const int l16  = lane & 15;
    const int z    = blockIdx.z;
    const int m0   = blockIdx.x * 256 + wave * 64;
    const int h    = blockIdx.y;                      // 64-wide N tile == one head

    const unsigned short* wt   = (z == 0) ? wqt : (z == 1) ? wkt : wvt;
    const float*          bias = (z == 0) ? bq  : (z == 1) ? bk  : bv;

    v8f acc[4][4];
    const v8f zro = {0.f, 0.f, 0.f, 0.f, 0.f, 0.f, 0.f, 0.f};
#pragma unroll
    for (int i = 0; i < 4; i++)
#pragma unroll
        for (int j = 0; j < 4; j++) acc[i][j] = zro;

    wave_gemm_64x64(xb + (size_t)m0 * DDM, wt + (size_t)h * HDIM * DDM, acc, l16, half);

    if (z < 2) {
        unsigned short* dst = (z == 0) ? qb : kb;
#pragma unroll
        for (int i = 0; i < 4; i++) {
            const int mb = m0 + i * 16 + half * 8;
            const int b  = mb >> 11;
            const int s0 = mb & (SSQ - 1);
#pragma unroll
            for (int j = 0; j < 4; j++) {
                const int   hd = j * 16 + l16;
                const float bv_ = bias[h * HDIM + hd];
                unsigned short* p = dst + (((size_t)(b * HHN + h) * SSQ + s0) * HDIM + hd);
#pragma unroll
                for (int r = 0; r < 8; r++)
                    p[(size_t)r * HDIM] = f2bf_bits(acc[i][j][r] + bv_);
            }
        }
    } else {
#pragma unroll
        for (int i = 0; i < 4; i++) {
            const int mb = m0 + i * 16 + half * 8;
            const int b  = mb >> 11;
            const int s0 = mb & (SSQ - 1);
#pragma unroll
            for (int j = 0; j < 4; j++) {
                const int   hd = j * 16 + l16;
                const float bv_ = bias[h * HDIM + hd];
                v4u pk;
#pragma unroll
                for (int q = 0; q < 4; q++)
                    pk[q] = (unsigned int)f2bf_bits(acc[i][j][2 * q] + bv_) |
                            ((unsigned int)f2bf_bits(acc[i][j][2 * q + 1] + bv_) << 16);
                *(v4u*)(vtb + ((size_t)(b * HHN + h) * HDIM + hd) * SSQ + s0) = pk;
            }
        }
    }
}

// ---------------------------------------------------------------------------
// Flash attention (causal, online softmax).  grid = (B*H, S/128), block = 256
// (8 waves).  Each wave owns 16 query rows; iterates 32-key tiles to the
// diagonal.  Scores: 4 WMMA / 32 keys.  P is transposed into the A-fragment
// layout through a 1KB/wave LDS bounce (same-wave LDS is in-order; explicit
// s_wait_dscnt 0 makes the cross-lane handoff visible).  P*V: 4 WMMA.
// ---------------------------------------------------------------------------
__global__ __launch_bounds__(256) void k_attn(const unsigned short* __restrict__ qb,
                                              const unsigned short* __restrict__ kb,
                                              const unsigned short* __restrict__ vtb,
                                              unsigned short* __restrict__ ctxb) {
    __shared__ __align__(16) unsigned short sp[8][16 * 32];

    const int wave = threadIdx.x >> 5;
    const int lane = threadIdx.x & 31;
    const int half = lane >> 4;
    const int l16  = lane & 15;
    const int bh   = blockIdx.x;                 // b*H + h
    const int q0   = blockIdx.y * 128 + wave * 16;

    const unsigned short* qbase = qb  + (size_t)bh * SSQ * HDIM;
    const unsigned short* kbase = kb  + (size_t)bh * SSQ * HDIM;
    const unsigned short* vbase = vtb + (size_t)bh * HDIM * SSQ;

    // Q fragments (hd 0..31 and 32..63), loaded once.
    FragBf qf[2];
    {
        const unsigned short* p = qbase + (size_t)(q0 + l16) * HDIM + half * 8;
        qf[0].u[0] = *(const v4u*)(p);
        qf[0].u[1] = *(const v4u*)(p + 16);
        qf[1].u[0] = *(const v4u*)(p + 32);
        qf[1].u[1] = *(const v4u*)(p + 48);
    }

    const v8f zro = {0.f, 0.f, 0.f, 0.f, 0.f, 0.f, 0.f, 0.f};
    v8f   o[4] = {zro, zro, zro, zro};
    float mrow[8], lrow[8];
#pragma unroll
    for (int r = 0; r < 8; r++) { mrow[r] = -1e30f; lrow[r] = 0.f; }

    const float scale = 0.125f;   // 1/sqrt(64)
    const float NEG   = -1e30f;

#pragma unroll 1
    for (int k0 = 0; k0 < q0 + 16; k0 += 32) {
        // ---- scores: S[16q x 32k] = Q @ K^T, scaled + causal-masked -------
        v8f s[2];
#pragma unroll
        for (int jn = 0; jn < 2; jn++) {
            FragBf kf0, kf1;
            const unsigned short* p = kbase + (size_t)(k0 + jn * 16 + l16) * HDIM + half * 8;
            kf0.u[0] = *(const v4u*)(p);
            kf0.u[1] = *(const v4u*)(p + 16);
            kf1.u[0] = *(const v4u*)(p + 32);
            kf1.u[1] = *(const v4u*)(p + 48);
            v8f a = zro;
            a = wmma_bf16(qf[0].v, kf0.v, a);
            a = wmma_bf16(qf[1].v, kf1.v, a);
            s[jn] = a;
        }
#pragma unroll
        for (int jn = 0; jn < 2; jn++) {
            const int kj = k0 + jn * 16 + l16;
#pragma unroll
            for (int r = 0; r < 8; r++) {
                const int qi = q0 + half * 8 + r;
                float v = s[jn][r] * scale;
                s[jn][r] = (kj > qi) ? NEG : v;
            }
        }
        // ---- online softmax ------------------------------------------------
#pragma unroll
        for (int r = 0; r < 8; r++) {
            float t = fmaxf(s[0][r], s[1][r]);
            t = fmaxf(t, SWZF(t, 1));
            t = fmaxf(t, SWZF(t, 2));
            t = fmaxf(t, SWZF(t, 4));
            t = fmaxf(t, SWZF(t, 8));
            const float mnew = fmaxf(mrow[r], t);
            const float corr = __expf(mrow[r] - mnew);
            mrow[r] = mnew;
            lrow[r] *= corr;
#pragma unroll
            for (int i = 0; i < 4; i++) o[i][r] *= corr;
            s[0][r] = __expf(s[0][r] - mnew);
            s[1][r] = __expf(s[1][r] - mnew);
            float u = s[0][r] + s[1][r];
            u += SWZF(u, 1);
            u += SWZF(u, 2);
            u += SWZF(u, 4);
            u += SWZF(u, 8);
            lrow[r] += u;
        }
        // ---- transpose P into A-fragment layout via LDS --------------------
#pragma unroll
        for (int jn = 0; jn < 2; jn++)
#pragma unroll
            for (int r = 0; r < 8; r++)
                sp[wave][(half * 8 + r) * 32 + jn * 16 + l16] = f2bf_bits(s[jn][r]);
        asm volatile("s_wait_dscnt 0x0" ::: "memory");
        FragBf pf;
        {
            const unsigned short* lp = &sp[wave][l16 * 32 + half * 8];
            pf.u[0] = *(const v4u*)(lp);
            pf.u[1] = *(const v4u*)(lp + 16);
        }
        // ---- O += P @ V  (V transposed: contiguous B-operand loads) --------
#pragma unroll
        for (int i = 0; i < 4; i++) {
            FragBf vf;
            const unsigned short* vp = vbase + (size_t)(i * 16 + l16) * SSQ + k0 + half * 8;
            vf.u[0] = *(const v4u*)(vp);
            vf.u[1] = *(const v4u*)(vp + 16);
            o[i] = wmma_bf16(pf.v, vf.v, o[i]);
        }
    }

    // ---- normalize and store ctx [B,S,D] bf16 ------------------------------
    const int b = bh / HHN;
    const int h = bh % HHN;
#pragma unroll
    for (int r = 0; r < 8; r++) {
        const int   sidx = q0 + half * 8 + r;
        const float inv  = 1.0f / lrow[r];
#pragma unroll
        for (int i = 0; i < 4; i++) {
            const int hd = i * 16 + l16;
            ctxb[((size_t)b * SSQ + sidx) * DDM + h * HDIM + hd] = f2bf_bits(o[i][r] * inv);
        }
    }
}

// ---------------------------------------------------------------------------
// Output projection: out = ctx @ Wo + bo   (f32 output)
// grid = (MMR/256, D/64), block = 128 (4 waves).
// ---------------------------------------------------------------------------
__global__ __launch_bounds__(128) void k_oproj(const unsigned short* __restrict__ ctxb,
                                               const unsigned short* __restrict__ wot,
                                               const float* __restrict__ bo,
                                               float* __restrict__ out) {
    const int wave = threadIdx.x >> 5;
    const int lane = threadIdx.x & 31;
    const int half = lane >> 4;
    const int l16  = lane & 15;
    const int m0   = blockIdx.x * 256 + wave * 64;
    const int n0   = blockIdx.y * 64;

    v8f acc[4][4];
    const v8f zro = {0.f, 0.f, 0.f, 0.f, 0.f, 0.f, 0.f, 0.f};
#pragma unroll
    for (int i = 0; i < 4; i++)
#pragma unroll
        for (int j = 0; j < 4; j++) acc[i][j] = zro;

    wave_gemm_64x64(ctxb + (size_t)m0 * DDM, wot + (size_t)n0 * DDM, acc, l16, half);

#pragma unroll
    for (int j = 0; j < 4; j++) {
        const int   n  = n0 + j * 16 + l16;
        const float bv = bo[n];
#pragma unroll
        for (int i = 0; i < 4; i++) {
            const int mb = m0 + i * 16 + half * 8;
#pragma unroll
            for (int r = 0; r < 8; r++)
                out[(size_t)(mb + r) * DDM + n] = acc[i][j][r] + bv;
        }
    }
}

// ---------------------------------------------------------------------------
extern "C" void kernel_launch(void* const* d_in, const int* in_sizes, int n_in,
                              void* d_out, int out_size, void* d_ws, size_t ws_size,
                              hipStream_t stream) {
    (void)in_sizes; (void)n_in; (void)out_size; (void)ws_size;

    const float* x  = (const float*)d_in[0];
    const float* Wq = (const float*)d_in[1];
    const float* bq = (const float*)d_in[2];
    const float* Wk = (const float*)d_in[3];
    const float* bk = (const float*)d_in[4];
    const float* Wv = (const float*)d_in[5];
    const float* bv = (const float*)d_in[6];
    const float* Wo = (const float*)d_in[7];
    const float* bo = (const float*)d_in[8];
    float* out = (float*)d_out;

    // workspace carve-up (bf16 buffers), 256B-aligned
    char*  ws  = (char*)d_ws;
    size_t off = 0;
    auto carve = [&](size_t bytes) -> unsigned short* {
        unsigned short* p = (unsigned short*)(ws + off);
        off += (bytes + 255) & ~(size_t)255;
        return p;
    };
    const size_t xdB = (size_t)MMR * DDM * 2;   // 6.29 MB
    const size_t wB  = (size_t)DDM * DDM * 2;   // 1.18 MB
    unsigned short* xb   = carve(xdB);
    unsigned short* wqt  = carve(wB);
    unsigned short* wkt  = carve(wB);
    unsigned short* wvt  = carve(wB);
    unsigned short* wot  = carve(wB);
    unsigned short* qb   = carve(xdB);
    unsigned short* kb   = carve(xdB);
    unsigned short* vtb  = carve(xdB);
    unsigned short* ctxb = carve(xdB);          // total ~34.5 MB

    // 1) precision conversions (bandwidth-trivial)
    {
        int n = MMR * DDM;
        k_f32_to_bf16<<<(n + 255) / 256, 256, 0, stream>>>(x, xb, n);
        int nw = DDM * DDM;
        int gw = (nw + 255) / 256;
        k_transpose_bf16<<<gw, 256, 0, stream>>>(Wq, wqt);
        k_transpose_bf16<<<gw, 256, 0, stream>>>(Wk, wkt);
        k_transpose_bf16<<<gw, 256, 0, stream>>>(Wv, wvt);
        k_transpose_bf16<<<gw, 256, 0, stream>>>(Wo, wot);
    }
    // 2) fused QKV projection (WMMA GEMM)
    k_qkv<<<dim3(MMR / 256, HHN, 3), 128, 0, stream>>>(xb, wqt, wkt, wvt,
                                                       bq, bk, bv, qb, kb, vtb);
    // 3) causal flash attention (WMMA)
    k_attn<<<dim3(BB * HHN, SSQ / 128), 256, 0, stream>>>(qb, kb, vtb, ctxb);
    // 4) output projection + bias (WMMA GEMM, f32 out)
    k_oproj<<<dim3(MMR / 256, DDM / 64), 128, 0, stream>>>(ctxb, wot, bo, out);
}
